// custom_gnn_3831110828329
// MI455X (gfx1250) — matrix operand order repaired and checked by reading, not compiled, not run on hardware
//
#include <hip/hip_runtime.h>
#include <math.h>

// ---------------------------------------------------------------------------
// Problem constants (from reference)
// ---------------------------------------------------------------------------
#define NN   50000
#define EE   200000
#define DD   384
#define HH   2
#define DHH  192
#define QKVC 1152        // 3*D: packed q|k|v columns
#define KK   384         // GEMM K (both GEMMs)
#define SB_STRIDE 392    // padded LDS row stride (elements): 784B rows, bank-conflict-free

typedef __attribute__((ext_vector_type(16))) __bf16          v16bf;
typedef __attribute__((ext_vector_type(8)))  float           v8f;
typedef __attribute__((ext_vector_type(8)))  unsigned short  ushort8;
typedef __attribute__((ext_vector_type(4)))  int             v4i;

typedef __attribute__((address_space(1))) v4i* glb_v4i_ptr;
typedef __attribute__((address_space(3))) v4i* lds_v4i_ptr;

union BFrag { v16bf v; ushort8 h[2]; };

#if __has_builtin(__builtin_amdgcn_global_load_async_to_lds_b128) && \
    __has_builtin(__builtin_amdgcn_s_wait_asynccnt)
#define HAVE_ASYNC_LDS 1
#else
#define HAVE_ASYNC_LDS 0
#endif

// round-to-nearest-even f32 -> bf16 bits
__device__ __forceinline__ unsigned short f32_to_bf16_bits(float f) {
    unsigned int x = __float_as_uint(f);
    unsigned int r = x + 0x7FFFu + ((x >> 16) & 1u);
    return (unsigned short)(r >> 16);
}

__device__ __forceinline__ void atomicMaxF32(float* addr, float v) {
    if (v >= 0.0f) atomicMax((int*)addr, __float_as_int(v));
    else           atomicMin((unsigned int*)addr, __float_as_uint(v));
}

// ---------------------------------------------------------------------------
// 1) elementwise f32 -> bf16 bits
// ---------------------------------------------------------------------------
__global__ void k_f32_to_bf16(const float* __restrict__ src,
                              unsigned short* __restrict__ dst, int n) {
    int i = blockIdx.x * blockDim.x + threadIdx.x;
    if (i < n) dst[i] = f32_to_bf16_bits(src[i]);
}

// ---------------------------------------------------------------------------
// 2) pack W(Q|K|V)^T  ->  BT[col*384 + k], col in [0,1152)
// ---------------------------------------------------------------------------
__global__ void k_pack_wqkvT(const float* __restrict__ WQ,
                             const float* __restrict__ WK,
                             const float* __restrict__ WV,
                             unsigned short* __restrict__ BT) {
    int i = blockIdx.x * blockDim.x + threadIdx.x;
    if (i >= QKVC * DD) return;
    int col = i / DD;
    int k   = i - col * DD;
    const float* W = (col < DD) ? WQ : (col < 2 * DD ? WK : WV);
    int c = col % DD;
    BT[i] = f32_to_bf16_bits(W[(size_t)k * DD + c]);
}

__global__ void k_pack_woT(const float* __restrict__ WO,
                           unsigned short* __restrict__ BT) {
    int i = blockIdx.x * blockDim.x + threadIdx.x;
    if (i >= DD * DD) return;
    int col = i / DD;
    int k   = i - col * DD;
    BT[i] = f32_to_bf16_bits(WO[(size_t)k * DD + col]);
}

// ---------------------------------------------------------------------------
// 3) init: smax=-inf, denom=0, agg=0
// ---------------------------------------------------------------------------
__global__ void k_init(float* __restrict__ smax, float* __restrict__ denom,
                       float* __restrict__ agg, int nSD, int nAgg) {
    int i = blockIdx.x * blockDim.x + threadIdx.x;
    if (i < nSD) { smax[i] = __uint_as_float(0xFF800000u); denom[i] = 0.0f; }
    if (i < nAgg) agg[i] = 0.0f;
}

// ---------------------------------------------------------------------------
// 4) bf16 WMMA GEMM, block-cooperative B staging in LDS.
//    C[M,Ncols](f32) = A[M,K=384](bf16) * B[384,Ncols](bf16)
//    B supplied transposed (BT[col*K + k], k-contiguous).
//    Block = 8 waves; block covers 128 rows x 64 cols. The 64-col B strip
//    (48 KB) is staged ONCE per block (async-to-LDS when available) and
//    reused by all 8 waves; each wave runs 12 K-steps of 4 WMMA tiles.
//    Fragment layouts per CDNA5 ISA 7.12.2 (16-bit A 16x32, B 32x16, f32 C).
// ---------------------------------------------------------------------------
__global__ __launch_bounds__(256)
void k_gemm_bf16_wmma(const unsigned short* __restrict__ A,
                      const unsigned short* __restrict__ BT,
                      float* __restrict__ C,
                      int M, int Ncols) {
    __shared__ unsigned short sB[64 * SB_STRIDE];        // 64 cols x 384 K (padded)

    const int lane = threadIdx.x & 31;
    const int wave = threadIdx.x >> 5;
    const int n0   = blockIdx.y * 64;                    // column strip
    const int mt   = blockIdx.x * 8 + wave;              // per-wave M tile

    // ---- cooperative stage: B strip -> LDS (16B chunks, 12 per thread) ----
    {
        const unsigned short* Bsrc = BT + (size_t)n0 * KK;
        const int nChunks = 64 * (KK / 8);               // 3072 x 16B
        for (int c = threadIdx.x; c < nChunks; c += blockDim.x) {
            const int col = c / (KK / 8);
            const int kc  = c - col * (KK / 8);
            const unsigned short* g = Bsrc + (size_t)col * KK + kc * 8;
            unsigned short*       l = sB + col * SB_STRIDE + kc * 8;
#if HAVE_ASYNC_LDS
            __builtin_amdgcn_global_load_async_to_lds_b128(
                (glb_v4i_ptr)g, (lds_v4i_ptr)l, 0, 0);
#else
            *(ushort8*)l = *(const ushort8*)g;
#endif
        }
    }
#if HAVE_ASYNC_LDS
    __builtin_amdgcn_s_wait_asynccnt(0);
#endif
    __syncthreads();

    if (mt * 16 < M) {                                   // wave-uniform guard
        const int m0 = mt * 16;

        const v8f zero = {0.f, 0.f, 0.f, 0.f, 0.f, 0.f, 0.f, 0.f};
        v8f acc[4] = {zero, zero, zero, zero};

        // A fragment (16x32 bf16): lanes<16 -> M=lane, K {0..7,16..23};
        // lanes>=16 -> M=lane-16, K {8..15,24..31}
        const int aRow = m0 + (lane & 15);
        const int aOff = (lane < 16) ? 0 : 8;
        const unsigned short* Arow = A + (size_t)aRow * KK;

        // B fragment (32x16 bf16): lanes<16 -> col=lane, K {0..7,16..23};
        // lanes>=16 -> col=lane-16, K {8..15,24..31}  (k-contiguous in sB row)
        const int bCol = lane & 15;
        const int bOff = (lane < 16) ? 0 : 8;

        for (int k0 = 0; k0 < KK; k0 += 32) {
            BFrag a;
            a.h[0] = *(const ushort8*)(Arow + k0 + aOff);
            a.h[1] = *(const ushort8*)(Arow + k0 + aOff + 16);
            if (k0 + 32 < KK)
                __builtin_prefetch(Arow + k0 + 32, 0, 3);    // global_prefetch_b8
#pragma unroll
            for (int t = 0; t < 4; ++t) {
                const unsigned short* bp =
                    sB + (t * 16 + bCol) * SB_STRIDE + k0 + bOff;
                BFrag b;
                b.h[0] = *(const ushort8*)bp;                // ds_load, 16B aligned
                b.h[1] = *(const ushort8*)(bp + 16);
                acc[t] = __builtin_amdgcn_wmma_f32_16x16x32_bf16(
                    false, a.v, false, b.v, (short)0, acc[t], false, false);
            }
        }

        // C layout: VGPR j -> row m0+j (lanes<16) / m0+8+j (lanes>=16)
        const int rbase = m0 + ((lane < 16) ? 0 : 8);
#pragma unroll
        for (int t = 0; t < 4; ++t) {
            const int col = n0 + t * 16 + (lane & 15);
#pragma unroll
            for (int j = 0; j < 8; ++j)
                C[(size_t)(rbase + j) * Ncols + col] = acc[t][j];
        }
    }
}

// ---------------------------------------------------------------------------
// 5) edge scores + segment-max (one wave per edge)
// ---------------------------------------------------------------------------
__global__ void k_edge_score(const float* __restrict__ qkv,
                             const int*   __restrict__ ei,
                             const float* __restrict__ ea,
                             const float* __restrict__ WE,
                             float* __restrict__ score,
                             float* __restrict__ smax) {
    const int lane = threadIdx.x & 31;
    const int e = blockIdx.x * (blockDim.x >> 5) + (threadIdx.x >> 5);
    if (e >= EE) return;
    const int src = ei[e];
    const int dst = ei[EE + e];
    const float a = ea[e];
    const float* q = qkv + (size_t)dst * QKVC;
    const float* k = qkv + (size_t)src * QKVC + DD;
    float s0 = 0.f, s1 = 0.f;
    for (int d = lane; d < DD; d += 32) {
        float kj = k[d] + a * WE[d];
        float p  = q[d] * kj;
        if (d < DHH) s0 += p; else s1 += p;
    }
    for (int m = 16; m >= 1; m >>= 1) {
        s0 += __shfl_xor(s0, m, 32);
        s1 += __shfl_xor(s1, m, 32);
    }
    if (lane == 0) {
        const float inv = 0.07216878364870323f;          // 1/sqrt(192)
        float sc0 = s0 * inv, sc1 = s1 * inv;
        score[e * 2 + 0] = sc0;
        score[e * 2 + 1] = sc1;
        atomicMaxF32(&smax[dst * 2 + 0], sc0);
        atomicMaxF32(&smax[dst * 2 + 1], sc1);
    }
}

// ---------------------------------------------------------------------------
// 6) exp(score - smax[dst]) and segment-sum (native f32 atomic add)
// ---------------------------------------------------------------------------
__global__ void k_edge_exp(const int* __restrict__ ei,
                           float* __restrict__ score,       // in: score, out: ex
                           const float* __restrict__ smax,
                           float* __restrict__ denom) {
    int i = blockIdx.x * blockDim.x + threadIdx.x;
    if (i >= EE * HH) return;
    int e = i >> 1, h = i & 1;
    int dst = ei[EE + e];
    float ex = __expf(score[i] - smax[dst * 2 + h]);
    score[i] = ex;
    atomicAdd(&denom[dst * 2 + h], ex);
}

// ---------------------------------------------------------------------------
// 7) weighted scatter-add aggregation (one wave per edge)
// ---------------------------------------------------------------------------
__global__ void k_edge_agg(const float* __restrict__ qkv,
                           const int*   __restrict__ ei,
                           const float* __restrict__ ea,
                           const float* __restrict__ WE,
                           const float* __restrict__ ex,
                           const float* __restrict__ denom,
                           float* __restrict__ agg) {
    const int lane = threadIdx.x & 31;
    const int e = blockIdx.x * (blockDim.x >> 5) + (threadIdx.x >> 5);
    if (e >= EE) return;
    const int src = ei[e];
    const int dst = ei[EE + e];
    const float a = ea[e];
    const float a0 = ex[e * 2 + 0] / (denom[dst * 2 + 0] + 1e-16f);
    const float a1 = ex[e * 2 + 1] / (denom[dst * 2 + 1] + 1e-16f);
    const float* v = qkv + (size_t)src * QKVC + 2 * DD;
    float* o = agg + (size_t)dst * DD;
    for (int d = lane; d < DD; d += 32) {
        float vj = v[d] + a * WE[d];
        float al = (d < DHH) ? a0 : a1;
        atomicAdd(&o[d], al * vj);
    }
}

// ---------------------------------------------------------------------------
// 8) residual + layernorm (one wave per row, 12 elems/lane)
// ---------------------------------------------------------------------------
__global__ void k_ln_residual(const float* __restrict__ y,
                              const float* __restrict__ emb,
                              const float* __restrict__ g,
                              const float* __restrict__ b,
                              float* __restrict__ out) {
    const int lane = threadIdx.x & 31;
    const int n = blockIdx.x * (blockDim.x >> 5) + (threadIdx.x >> 5);
    if (n >= NN) return;
    const float* yr = y   + (size_t)n * DD;
    const float* er = emb + (size_t)n * DD;
    float x[12];
    float s = 0.f;
#pragma unroll
    for (int i = 0; i < 12; ++i) {
        x[i] = yr[lane + i * 32] + er[lane + i * 32];
        s += x[i];
    }
    for (int m = 16; m >= 1; m >>= 1) s += __shfl_xor(s, m, 32);
    const float mu = s * (1.0f / DD);
    float vv = 0.f;
#pragma unroll
    for (int i = 0; i < 12; ++i) { float d = x[i] - mu; vv += d * d; }
    for (int m = 16; m >= 1; m >>= 1) vv += __shfl_xor(vv, m, 32);
    const float rstd = rsqrtf(vv * (1.0f / DD) + 1e-5f);
    float* orow = out + (size_t)n * DD;
#pragma unroll
    for (int i = 0; i < 12; ++i) {
        int d = lane + i * 32;
        orow[d] = (x[i] - mu) * rstd * g[d] + b[d];
    }
}

// ---------------------------------------------------------------------------
// launcher
// ---------------------------------------------------------------------------
static inline size_t align256(size_t x) { return (x + 255u) & ~(size_t)255u; }

extern "C" void kernel_launch(void* const* d_in, const int* in_sizes, int n_in,
                              void* d_out, int out_size, void* d_ws, size_t ws_size,
                              hipStream_t stream) {
    const float* emb = (const float*)d_in[0];
    const int*   ei  = (const int*)  d_in[1];
    const float* ea  = (const float*)d_in[2];
    const float* WQ  = (const float*)d_in[3];
    const float* WK  = (const float*)d_in[4];
    const float* WV  = (const float*)d_in[5];
    const float* WE  = (const float*)d_in[6];
    const float* WO  = (const float*)d_in[7];
    const float* lng = (const float*)d_in[8];
    const float* lnb = (const float*)d_in[9];
    float* out = (float*)d_out;

    char* ws = (char*)d_ws;
    size_t off = 0;
    unsigned short* embBF  = (unsigned short*)(ws + off); off = align256(off + (size_t)NN * DD * 2);
    unsigned short* WqkvT  = (unsigned short*)(ws + off); off = align256(off + (size_t)QKVC * DD * 2);
    unsigned short* WoT    = (unsigned short*)(ws + off); off = align256(off + (size_t)DD * DD * 2);
    float*          qkv    = (float*)        (ws + off); off = align256(off + (size_t)NN * QKVC * 4);
    float*          score  = (float*)        (ws + off); off = align256(off + (size_t)EE * HH * 4);
    float*          smax   = (float*)        (ws + off); off = align256(off + (size_t)NN * HH * 4);
    float*          denom  = (float*)        (ws + off); off = align256(off + (size_t)NN * HH * 4);
    float*          agg    = (float*)        (ws + off); off = align256(off + (size_t)NN * DD * 4);
    unsigned short* aggBF  = (unsigned short*)(ws + off); off = align256(off + (size_t)NN * DD * 2);
    float*          y2     = (float*)        (ws + off); off = align256(off + (size_t)NN * DD * 4);
    (void)ws_size; (void)in_sizes; (void)n_in; (void)out_size;

    const int nEmb   = NN * DD;                             // 19.2M
    const int mTiles = NN / 16;                             // 3125
    const int mBlks  = (mTiles + 7) / 8;                    // 391 (last partial)

    // --- stage 0: precision conversion / weight packing -------------------
    k_f32_to_bf16<<<(nEmb + 255) / 256, 256, 0, stream>>>(emb, embBF, nEmb);
    k_pack_wqkvT<<<(QKVC * DD + 255) / 256, 256, 0, stream>>>(WQ, WK, WV, WqkvT);
    k_pack_woT<<<(DD * DD + 255) / 256, 256, 0, stream>>>(WO, WoT);
    k_init<<<(nEmb + 255) / 256, 256, 0, stream>>>(smax, denom, agg, NN * HH, nEmb);

    // --- stage 1: fused QKV projection (WMMA bf16, LDS-staged B) ----------
    {
        dim3 grid(mBlks, QKVC / 64);                        // 391 x 18
        k_gemm_bf16_wmma<<<grid, 256, 0, stream>>>(embBF, WqkvT, qkv, NN, QKVC);
    }

    // --- stage 2: edge attention ------------------------------------------
    k_edge_score<<<(EE + 7) / 8, 256, 0, stream>>>(qkv, ei, ea, WE, score, smax);
    k_edge_exp<<<(EE * HH + 255) / 256, 256, 0, stream>>>(ei, score, smax, denom);
    k_edge_agg<<<(EE + 7) / 8, 256, 0, stream>>>(qkv, ei, ea, WE, score, denom, agg);

    // --- stage 3: output projection (WMMA bf16) + residual LN -------------
    k_f32_to_bf16<<<(nEmb + 255) / 256, 256, 0, stream>>>(agg, aggBF, nEmb);
    {
        dim3 grid(mBlks, DD / 64);                          // 391 x 6
        k_gemm_bf16_wmma<<<grid, 256, 0, stream>>>(aggBF, WoT, y2, NN, DD);
    }
    k_ln_residual<<<(NN + 7) / 8, 256, 0, stream>>>(y2, emb, lng, lnb, out);
}
